// MultiHeadAttention_50955492000010
// MI455X (gfx1250) — matrix operand order repaired and checked by reading, not compiled, not run on hardware
//
#include <hip/hip_runtime.h>
#include <hip/hip_bf16.h>

// ---------------------------------------------------------------------------
// MI455X fused multi-head attention, f16 WMMA everywhere, f32 accumulate.
//   d = 1024, H = 16, dk = 64, B = 4, S = 1024
// Pipeline:
//   cvt      : query/key/value, Wq/Wk/Wv/Wo  f32 -> f16   (one pass, HBM-bound)
//   gemm(Q)  : Qp  = f16( (query @ Wq^T) * log2e/8 )      [B*S, D] row-major
//   gemm(K)  : Kp  = f16(  key   @ Wk^T )                 [B*S, D] row-major
//   gemm(V)  : Vt  = f16( value  @ Wv^T ) transposed  ->  [B, H, dk, S]
//   flash    : Aout= f16( softmax2(Qp Kp^T, causal) @ V ) [B*S, D] row-major
//   gemm(O)  : out = f32( Aout @ Wo^T + bo )              [B*S, D]
// GEMM inner loop is pure f16 global_load_b128 + v_wmma (no cvt in loop).
// ---------------------------------------------------------------------------

typedef __attribute__((ext_vector_type(16))) _Float16 v16h;
typedef __attribute__((ext_vector_type(8)))  _Float16 v8h;
typedef __attribute__((ext_vector_type(8)))  float    v8f;
typedef __attribute__((ext_vector_type(4)))  float    v4f;

#define S_LEN 1024
#define D_MODEL 1024
#define NHEAD 16
#define DK 64

static __device__ __forceinline__ v8f wmma16x16x32(v16h a, v16h b, v8f c) {
  return __builtin_amdgcn_wmma_f32_16x16x32_f16(
      /*neg_a=*/false, a, /*neg_b=*/false, b,
      /*c_mod=*/(short)0, c, /*reuse_a=*/false, /*reuse_b=*/false);
}

static __device__ __forceinline__ v8f zero8() {
  v8f z;
#pragma unroll
  for (int i = 0; i < 8; ++i) z[i] = 0.0f;
  return z;
}

static __device__ __forceinline__ v16h comb(v8h lo, v8h hi2) {
  v16h r;
#pragma unroll
  for (int i = 0; i < 8; ++i) { r[i] = lo[i]; r[i + 8] = hi2[i]; }
  return r;
}

// A-fragment (16x32 f16, MxK): lane L<16 holds row L, K = 0..7 & 16..23;
// lane L>=16 holds row L-16, K = 8..15 & 24..31.  p = row start + k0.
static __device__ __forceinline__ v16h load_a_f16(const _Float16* p, int hi) {
  return comb(*(const v8h*)(p + hi * 8), *(const v8h*)(p + 16 + hi * 8));
}
// B-fragment (32x16 f16, KxN): lane L<16 holds col L, K = 0..7 & 16..23;
// lane L>=16 holds col L-16, K = 8..15 & 24..31.  p = col-source row start + k0.
static __device__ __forceinline__ v16h load_b_f16(const _Float16* p, int hi) {
  return comb(*(const v8h*)(p + hi * 8), *(const v8h*)(p + 16 + hi * 8));
}

// ---------------------------------------------------------------------------
// Elementwise f32 -> f16 (8 elements / thread, 32B-aligned vector ld/st)
// ---------------------------------------------------------------------------
__global__ __launch_bounds__(256) void cvt_f32_to_f16_kernel(
    const float* __restrict__ in, _Float16* __restrict__ out, int n8) {
  const int i = blockIdx.x * blockDim.x + threadIdx.x;
  if (i >= n8) return;
  const v4f* p = (const v4f*)(in + (size_t)i * 8);
  v8h o;
#pragma unroll
  for (int j = 0; j < 4; ++j) { o[j] = (_Float16)p[0][j]; o[j + 4] = (_Float16)p[1][j]; }
  *(v8h*)(out + (size_t)i * 8) = o;
}

// ---------------------------------------------------------------------------
// GEMM: out = A @ W^T, all-f16 operands (A [M,K], W [N,K] row-major).
// Block = 8 waves, tile 256x64: wave w owns rows [m0, m0+32) (2 A-frags),
// all 64 cols (4 B-frags reused across both row tiles) -> 8 WMMA / K-step.
// mode 0: f16 rowmajor            mode 1: f16 rowmajor * scale
// mode 2: f16 scattered to Vt[B,H,dk,S]   mode 3: f32 + bias to d_out
// ---------------------------------------------------------------------------
__global__ __launch_bounds__(256) void gemm_wmma_kernel(
    const _Float16* __restrict__ A, const _Float16* __restrict__ W,
    const float* __restrict__ bias, void* __restrict__ outp,
    int mode, float scale, int N, int K) {
  const int lane = threadIdx.x & 31;
  const int wave = threadIdx.x >> 5;
  const int lr = lane & 15;
  const int hi = lane >> 4;
  const int m0 = blockIdx.x * 256 + wave * 32;
  const int n0 = blockIdx.y * 64;

  v8f c[8];                       // c[rt*4 + t]: row tile rt (0,1), col tile t
#pragma unroll
  for (int i = 0; i < 8; ++i) c[i] = zero8();

  const _Float16* arow0 = A + (size_t)(m0 + lr) * K;
  const _Float16* arow1 = A + (size_t)(m0 + 16 + lr) * K;
  for (int k0 = 0; k0 < K; k0 += 32) {
    const v16h a0 = load_a_f16(arow0 + k0, hi);
    const v16h a1 = load_a_f16(arow1 + k0, hi);
#pragma unroll
    for (int t = 0; t < 4; ++t) {
      const v16h b = load_b_f16(W + (size_t)(n0 + t * 16 + lr) * K + k0, hi);
      c[t]     = wmma16x16x32(a0, b, c[t]);
      c[4 + t] = wmma16x16x32(a1, b, c[4 + t]);
    }
  }

#pragma unroll
  for (int rt = 0; rt < 2; ++rt)
#pragma unroll
    for (int t = 0; t < 4; ++t)
#pragma unroll
      for (int r = 0; r < 8; ++r) {
        const int mm = m0 + rt * 16 + r + 8 * hi;
        const int nn = n0 + t * 16 + lr;
        const float v = c[rt * 4 + t][r];
        if (mode == 3) {
          ((float*)outp)[(size_t)mm * N + nn] = v + bias[nn];
        } else if (mode == 2) {
          const int bb = mm >> 10, sI = mm & 1023;   // (batch, seq)
          const int hh = nn >> 6,  dd = nn & 63;     // (head, dk)
          ((_Float16*)outp)[(((size_t)(bb * NHEAD + hh) * DK + dd) << 10) + sI] =
              (_Float16)v;
        } else if (mode == 1) {
          ((_Float16*)outp)[(size_t)mm * N + nn] = (_Float16)(v * scale);
        } else {
          ((_Float16*)outp)[(size_t)mm * N + nn] = (_Float16)v;
        }
      }
}

// ---------------------------------------------------------------------------
// Flash attention (base-2 exp domain; log2e/sqrt(dk) pre-folded into Qp).
// One wave = one 16-query tile of one (b,h); 8 waves/block; causal chunks
// of 32 keys. Per-wave private LDS does the P (C-layout -> A-layout) turn.
// No block barriers: waves have different trip counts.
// ---------------------------------------------------------------------------
__global__ __launch_bounds__(256) void flash_attn_kernel(
    const _Float16* __restrict__ Qp, const _Float16* __restrict__ Kp,
    const _Float16* __restrict__ Vt, _Float16* __restrict__ Aout) {
  const int lane = threadIdx.x & 31;
  const int wave = threadIdx.x >> 5;
  const int lr = lane & 15;
  const int hi = lane >> 4;
  const int blk = blockIdx.x;              // b*128 + h*8 + qb
  const int qb = blk & 7;
  const int h = (blk >> 3) & (NHEAD - 1);
  const int b = blk >> 7;
  const int q0 = qb * 128 + wave * 16;

  __shared__ _Float16 lds[8][16 * 40];     // per-wave 16 rows, stride 40 halfs
  _Float16* myl = &lds[wave][0];

  const _Float16* qbase = Qp + (size_t)b * S_LEN * D_MODEL + h * DK;
  const _Float16* kbase = Kp + (size_t)b * S_LEN * D_MODEL + h * DK;
  const _Float16* vbase = Vt + (size_t)(b * NHEAD + h) * DK * S_LEN;

  // Q A-fragments for dk chunks [0,32) and [32,64)
  const _Float16* qrow = qbase + (size_t)(q0 + lr) * D_MODEL;
  const v16h aq0 = load_a_f16(qrow, hi);
  const v16h aq1 = load_a_f16(qrow + 32, hi);

  v8f o[4];
#pragma unroll
  for (int t = 0; t < 4; ++t) o[t] = zero8();
  float m[8], s[8];
#pragma unroll
  for (int r = 0; r < 8; ++r) { m[r] = -3.0e38f; s[r] = 0.0f; }

  for (int kc = 0; kc <= q0 + 15; kc += 32) {
    // ---- scores: two 16-key tiles, contraction over dk=64 (2 WMMA each)
    const _Float16* kr0 = kbase + (size_t)(kc + lr) * D_MODEL;
    const _Float16* kr1 = kbase + (size_t)(kc + 16 + lr) * D_MODEL;
    v8f c0 = wmma16x16x32(aq1, load_b_f16(kr0 + 32, hi),
             wmma16x16x32(aq0, load_b_f16(kr0, hi), zero8()));
    v8f c1 = wmma16x16x32(aq1, load_b_f16(kr1 + 32, hi),
             wmma16x16x32(aq0, load_b_f16(kr1, hi), zero8()));

    // ---- causal mask (C-layout: row q0+r+8*hi, col kc+{0,16}+lr)
#pragma unroll
    for (int r = 0; r < 8; ++r) {
      const int q = q0 + r + 8 * hi;
      if (kc + lr > q)      c0[r] = -3.0e38f;
      if (kc + 16 + lr > q) c1[r] = -3.0e38f;
    }

    // ---- online softmax stats (base-2), rows duplicated per 16-lane group
    float f[8], p0[8], p1[8];
#pragma unroll
    for (int r = 0; r < 8; ++r) {
      float v = fmaxf(c0[r], c1[r]);
      v = fmaxf(v, __shfl_xor(v, 1, 16));
      v = fmaxf(v, __shfl_xor(v, 2, 16));
      v = fmaxf(v, __shfl_xor(v, 4, 16));
      v = fmaxf(v, __shfl_xor(v, 8, 16));
      const float mn = fmaxf(m[r], v);
      f[r] = exp2f(m[r] - mn);
      m[r] = mn;
      p0[r] = exp2f(c0[r] - mn);
      p1[r] = exp2f(c1[r] - mn);
      float rs = p0[r] + p1[r];
      rs += __shfl_xor(rs, 1, 16);
      rs += __shfl_xor(rs, 2, 16);
      rs += __shfl_xor(rs, 4, 16);
      rs += __shfl_xor(rs, 8, 16);
      s[r] = s[r] * f[r] + rs;
    }
#pragma unroll
    for (int t = 0; t < 4; ++t)
#pragma unroll
      for (int r = 0; r < 8; ++r) o[t][r] *= f[r];

    // ---- P: C-layout -> A-layout via per-wave LDS (same-wave DS in order)
#pragma unroll
    for (int r = 0; r < 8; ++r) {
      const int row = r + 8 * hi;
      myl[row * 40 + lr]      = (_Float16)p0[r];
      myl[row * 40 + 16 + lr] = (_Float16)p1[r];
    }
    const _Float16* pr = myl + lr * 40;
    const v16h pa = comb(*(const v8h*)(pr + hi * 8),
                         *(const v8h*)(pr + 16 + hi * 8));

    // ---- O += P @ V   (B-frags contiguous thanks to Vt[B,H,dk,S])
#pragma unroll
    for (int t = 0; t < 4; ++t) {
      const _Float16* vr = vbase + (size_t)(t * 16 + lr) * S_LEN + kc;
      o[t] = wmma16x16x32(pa, load_b_f16(vr, hi), o[t]);
    }
  }

  // ---- normalize and emit f16 [B,S,H*dk]
  float inv[8];
#pragma unroll
  for (int r = 0; r < 8; ++r) inv[r] = 1.0f / s[r];
#pragma unroll
  for (int t = 0; t < 4; ++t)
#pragma unroll
    for (int r = 0; r < 8; ++r) {
      const int q = q0 + r + 8 * hi;
      const int col = h * DK + t * 16 + lr;
      Aout[(size_t)(b * S_LEN + q) * D_MODEL + col] = (_Float16)(o[t][r] * inv[r]);
    }
}

// ---------------------------------------------------------------------------
extern "C" void kernel_launch(void* const* d_in, const int* in_sizes, int n_in,
                              void* d_out, int out_size, void* d_ws, size_t ws_size,
                              hipStream_t stream) {
  const float* query = (const float*)d_in[0];
  const float* key   = (const float*)d_in[1];
  const float* value = (const float*)d_in[2];
  // d_in[3] = causal mask: enforced analytically in flash_attn_kernel
  const float* Wq = (const float*)d_in[4];
  const float* Wk = (const float*)d_in[5];
  const float* Wv = (const float*)d_in[6];
  const float* Wo = (const float*)d_in[7];
  const float* bo = (const float*)d_in[8];

  const int M = 4 * S_LEN;                 // 4096 rows
  const int N = D_MODEL, K = D_MODEL;
  const int nAct = M * D_MODEL;            // 4M elements
  const int nWt  = D_MODEL * D_MODEL;      // 1M elements

  char* ws = (char*)d_ws;
  const size_t MB = (size_t)1 << 20;
  _Float16* Xq   = (_Float16*)(ws);             // 8 MB  f16 activations
  _Float16* Xk   = (_Float16*)(ws + 8 * MB);    // 8 MB
  _Float16* Xv   = (_Float16*)(ws + 16 * MB);   // 8 MB
  _Float16* Wq16 = (_Float16*)(ws + 24 * MB);   // 2 MB  f16 weights
  _Float16* Wk16 = (_Float16*)(ws + 26 * MB);   // 2 MB
  _Float16* Wv16 = (_Float16*)(ws + 28 * MB);   // 2 MB
  _Float16* Wo16 = (_Float16*)(ws + 30 * MB);   // 2 MB
  _Float16* Qp   = (_Float16*)(ws + 32 * MB);   // 8 MB  projections
  _Float16* Kp   = (_Float16*)(ws + 40 * MB);   // 8 MB
  _Float16* Vt   = (_Float16*)(ws + 48 * MB);   // 8 MB  [B,H,dk,S]
  _Float16* Aout = (_Float16*)(ws + 56 * MB);   // 8 MB

  // ---- one-shot f32 -> f16 conversion (bandwidth-bound, ~64 MB read)
  const int actBlk = (nAct / 8 + 255) / 256;
  const int wtBlk  = (nWt  / 8 + 255) / 256;
  cvt_f32_to_f16_kernel<<<actBlk, 256, 0, stream>>>(query, Xq, nAct / 8);
  cvt_f32_to_f16_kernel<<<actBlk, 256, 0, stream>>>(key,   Xk, nAct / 8);
  cvt_f32_to_f16_kernel<<<actBlk, 256, 0, stream>>>(value, Xv, nAct / 8);
  cvt_f32_to_f16_kernel<<<wtBlk, 256, 0, stream>>>(Wq, Wq16, nWt / 8);
  cvt_f32_to_f16_kernel<<<wtBlk, 256, 0, stream>>>(Wk, Wk16, nWt / 8);
  cvt_f32_to_f16_kernel<<<wtBlk, 256, 0, stream>>>(Wv, Wv16, nWt / 8);
  cvt_f32_to_f16_kernel<<<wtBlk, 256, 0, stream>>>(Wo, Wo16, nWt / 8);

  const float qscale = 0.125f * 1.44269504088896f;   // log2(e)/sqrt(dk)

  dim3 gg(M / 256, N / 64), gb(256);
  gemm_wmma_kernel<<<gg, gb, 0, stream>>>(Xq, Wq16, nullptr, Qp, 1, qscale, N, K);
  gemm_wmma_kernel<<<gg, gb, 0, stream>>>(Xk, Wk16, nullptr, Kp, 0, 1.0f,   N, K);
  gemm_wmma_kernel<<<gg, gb, 0, stream>>>(Xv, Wv16, nullptr, Vt, 2, 1.0f,   N, K);

  flash_attn_kernel<<<dim3(4 * NHEAD * (S_LEN / 128)), gb, 0, stream>>>(Qp, Kp, Vt, Aout);

  gemm_wmma_kernel<<<gg, gb, 0, stream>>>(Aout, Wo16, bo, d_out, 3, 1.0f, N, K);
}
